// Corr3D_2662879723529
// MI455X (gfx1250) — matrix operand order repaired and checked by reading, not compiled, not run on hardware
//
#include <hip/hip_runtime.h>

// CDNA5 / gfx1250 3D correlation via banded-Gram WMMA, 2x2 spatially tiled,
// with async LDS->memory output stores (ASYNCcnt path).
//
// out[b, (dy*9+dx)*9+dz, h, w, d] = sum_c in1[b,c,h,w,d] * in2_pad[b,c,h+dy-4,w+dx-4,d+dz-4]
//
// Block owns a 2x2 (h,w) patch. For each of the 10x10 neighbor lines (y2,x2),
// each paired position's 9x32 (dz,d) slab is the 9-diagonal band of
// G = W2^T * W1 (48x32 padded). 4 16x16 tiles touch the band -> 4 waves,
// one tile each, K=64 via two v_wmma_f32_16x16x32_f16. Bands are staged in LDS
// and written out with global_store_async_from_lds_b128 (16B/lane, 72 lanes).

typedef __attribute__((ext_vector_type(16))) _Float16 v16h;
typedef __attribute__((ext_vector_type(8)))  float    v8f;

#define SZ   32            // H = W = D
#define CH   64            // channels (reduction K)
#define KD   9             // 2*MAX_DIS+1
#define PAD  4             // MAX_DIS
#define Q    729           // KD^3
#define CST  (SZ*SZ*SZ)    // channel stride (32768 elements)
#define LSTR 66            // LDS row stride in halves (bank-friendly)
#define BSTR (KD*SZ + 128) // 288 band entries + 128 per-wave dump slots (x4B = 16B-mult)

// LDS -> global, 16 bytes per active lane, tracked by ASYNCcnt.
__device__ __forceinline__ void async_store_b128(const float* gptr, const float* lptr) {
    unsigned long long ga = (unsigned long long)(uintptr_t)gptr;
    unsigned           la = (unsigned)(uintptr_t)lptr;   // low 32b = LDS offset
    asm volatile("global_store_async_from_lds_b128 %0, %1, off"
                 :: "v"(ga), "v"(la) : "memory");
}
__device__ __forceinline__ void wait_asynccnt0() {
    asm volatile("s_wait_asynccnt 0x0" ::: "memory");
}

__global__ __launch_bounds__(128)
void corr3d_wmma(const float* __restrict__ in1,
                 const float* __restrict__ in2,
                 float* __restrict__ out)
{
    __shared__ _Float16 ln1T[4][SZ * LSTR];          // 4 in1 lines [z][c], f16
    __shared__ _Float16 ln2T[48 * LSTR];             // in2 line [p][c], p=z+4
    __shared__ __align__(16) float band[4][BSTR];    // band staging + dump
    __shared__ __align__(16) float zbuf[KD * SZ];    // constant zero slab

    const int tid = threadIdx.x;             // 128 threads = 4 waves (wave32)
    const int blk = blockIdx.x;              // 2*16*16 = 512 blocks
    const int b   = blk >> 8;
    const int h0  = ((blk >> 4) & 15) * 2;
    const int w0  = (blk & 15) * 2;

    const int lane = tid & 31;
    const int wave = tid >> 5;
    const int rt   = (wave + 1) >> 1;        // G row-tile: 0,1,1,2
    const int ct   = wave >> 1;              // G col-tile: 0,0,1,1
    const int hi   = lane >> 4;
    const int ln   = lane & 15;
    const int p    = rt * 16 + ln;           // A: lane's G row (padded z2)
    const int z1   = ct * 16 + ln;           // B: lane's G col (in1 z)
    const int dump = KD * SZ + wave * 32 + lane;   // branch-free scatter sink

    // zero padded slab once (rows p<4, p>=36 stay zero) + zero output buffer
    for (int i = tid; i < 48 * LSTR; i += 128) ln2T[i] = (_Float16)0.f;
    for (int i = tid; i < KD * SZ; i += 128) zbuf[i] = 0.f;

    // stage the 4 in1 lines as f16, z-major
    for (int i = tid; i < 4 * 512; i += 128) {
        const int pp = i >> 9;
        const int r  = i & 511;
        const int c  = r >> 3;
        const int z  = (r & 7) * 4;
        const int h  = h0 + (pp >> 1), w = w0 + (pp & 1);
        const float4 v = *(const float4*)(in1 + b*(CH*CST) + h*(SZ*SZ) + w*SZ
                                          + c*CST + z);
        ln1T[pp][(z + 0) * LSTR + c] = (_Float16)v.x;
        ln1T[pp][(z + 1) * LSTR + c] = (_Float16)v.y;
        ln1T[pp][(z + 2) * LSTR + c] = (_Float16)v.z;
        ln1T[pp][(z + 3) * LSTR + c] = (_Float16)v.w;
    }
    __syncthreads();

    // per-lane 16B chunk of a 9x32 slab: row dz = tid>>3, 4 floats at (tid&7)*4
    const int odz  = tid >> 3;
    const int ocol = (tid & 7) * 4;
    const bool owriter = (tid < (KD * SZ) / 4);      // 72 active lanes

    for (int line = 0; line < 100; ++line) {
        const int iy = line / 10, ix = line - iy * 10;
        const int y2 = h0 + iy - PAD, x2 = w0 + ix - PAD;
        const bool lineValid = ((unsigned)y2 < SZ) && ((unsigned)x2 < SZ);

        // which patch positions pair with this line (all block-uniform)
        bool   pvA[4];
        size_t obA[4];
        #pragma unroll
        for (int pp = 0; pp < 4; ++pp) {
            const int ph = pp >> 1, pw = pp & 1;
            const int dy = iy - ph, dx = ix - pw;
            pvA[pp] = ((unsigned)dy < KD) && ((unsigned)dx < KD);
            obA[pp] = ((size_t)(b * Q + (dy * KD + dx) * KD)) * CST
                    + (size_t)(h0 + ph) * (SZ * SZ) + (size_t)(w0 + pw) * SZ;
        }

        // out-of-range neighbor line: async-store the constant zero slab
        if (!lineValid) {
            #pragma unroll
            for (int pp = 0; pp < 4; ++pp)
                if (pvA[pp] && owriter)
                    async_store_b128(out + obA[pp] + (size_t)odz * CST + ocol,
                                     &zbuf[tid * 4]);
            continue;   // zbuf is never modified -> no wait needed here
        }

        // prefetch next neighbor line (gfx1250 global_prefetch_b8)
        if (line + 1 < 100) {
            const int niy = (line + 1) / 10, nix = (line + 1) - niy * 10;
            const int ny = h0 + niy - PAD, nx = w0 + nix - PAD;
            if ((unsigned)ny < SZ && (unsigned)nx < SZ)
                __builtin_prefetch(in2 + b*(CH*CST) + ny*(SZ*SZ) + nx*SZ
                                   + (tid >> 1) * CST + (tid & 1) * 16, 0, 1);
        }

        // stage in2 line (b,:,y2,x2,:) as f16 into padded rows p = z+4
        const int base2 = b*(CH*CST) + y2*(SZ*SZ) + x2*SZ;
        for (int i = tid; i < 512; i += 128) {
            const int c = i >> 3, z = (i & 7) * 4;
            const float4 v = *(const float4*)(in2 + base2 + c*CST + z);
            ln2T[(z + PAD + 0) * LSTR + c] = (_Float16)v.x;
            ln2T[(z + PAD + 1) * LSTR + c] = (_Float16)v.y;
            ln2T[(z + PAD + 2) * LSTR + c] = (_Float16)v.z;
            ln2T[(z + PAD + 3) * LSTR + c] = (_Float16)v.w;
        }
        __syncthreads();   // (A) staging visible

        // A fragments depend only on the line -> load once, reuse for 4 pairs.
        union F { unsigned int u[8]; v16h h; };
        F A0, A1;
        #pragma unroll
        for (int j = 0; j < 8; ++j) {
            const int ca = (j < 4) ? (2*j + 8*hi) : (16 + 2*(j - 4) + 8*hi);
            A0.u[j] = *(const unsigned int*)&ln2T[p * LSTR + ca];
            A1.u[j] = *(const unsigned int*)&ln2T[p * LSTR + 32 + ca];
        }

        #pragma unroll
        for (int pp = 0; pp < 4; ++pp) {
            if (!pvA[pp]) continue;            // uniform -> EXEC stays full
            F B0, B1;
            #pragma unroll
            for (int j = 0; j < 8; ++j) {
                const int cbx = (j < 4) ? (2*j + 8*hi) : (16 + 2*(j - 4) + 8*hi);
                B0.u[j] = *(const unsigned int*)&ln1T[pp][z1 * LSTR + cbx];
                B1.u[j] = *(const unsigned int*)&ln1T[pp][z1 * LSTR + 32 + cbx];
            }
            v8f acc = {0.f, 0.f, 0.f, 0.f, 0.f, 0.f, 0.f, 0.f};
            acc = __builtin_amdgcn_wmma_f32_16x16x32_f16(
                      false, A0.h, false, B0.h, (short)0, acc, false, false);
            acc = __builtin_amdgcn_wmma_f32_16x16x32_f16(
                      false, A1.h, false, B1.h, (short)0, acc, false, false);

            // branch-free band scatter: dz = z2p - z1 in [0,9) else dump slot
            #pragma unroll
            for (int r = 0; r < 8; ++r) {
                const int dz  = (rt - ct) * 16 + r + 8 * hi - ln;
                const int idx = ((unsigned)dz < KD) ? (dz * SZ + z1) : dump;
                band[pp][idx] = acc[r];
            }
        }
        __syncthreads();   // (B) bands complete (barrier + dscnt fence)

        // async DMA the 4 band slabs straight from LDS (no VGPR round-trip)
        #pragma unroll
        for (int pp = 0; pp < 4; ++pp)
            if (pvA[pp] && owriter)
                async_store_b128(out + obA[pp] + (size_t)odz * CST + ocol,
                                 &band[pp][tid * 4]);

        wait_asynccnt0();  // DMA reads of band done before it is rewritten
        __syncthreads();   // (C) protect band/ln2T from next line's writers
    }
}

extern "C" void kernel_launch(void* const* d_in, const int* in_sizes, int n_in,
                              void* d_out, int out_size, void* d_ws, size_t ws_size,
                              hipStream_t stream) {
    const float* in1 = (const float*)d_in[0];
    const float* in2 = (const float*)d_in[1];
    float* out = (float*)d_out;
    // One block per (b, 2x2 hw-patch): 2*16*16 = 512 blocks, 4 waves each.
    corr3d_wmma<<<2 * 16 * 16, 128, 0, stream>>>(in1, in2, out);
}
